// LinearChainCRF_31473520345408
// MI455X (gfx1250) — matrix operand order repaired and checked by reading, not compiled, not run on hardware
//
#include <hip/hip_runtime.h>
#include <math.h>

typedef __attribute__((ext_vector_type(16))) _Float16 v16h;
typedef __attribute__((ext_vector_type(8)))  float    v8f;

#define NEGV  (-10000000000.0f)
#define Bsz   64
#define Tsz   512
#define Lsz   64
// START = 0, END = 63

// Raw-rate transcendentals: inputs to FAST_LOG are clamped >= 1e-30 (normal),
// so the OCML denormal fix-up path (cmp/cndmask/ldexp around v_log_f32) is dead
// weight on the serial critical path — use the bare instructions.
#if __has_builtin(__builtin_amdgcn_exp2f)
#define FAST_EXP(x) __builtin_amdgcn_exp2f((x) * 1.44269504088896340736f)  // v_exp_f32
#else
#define FAST_EXP(x) __expf(x)
#endif
#if __has_builtin(__builtin_amdgcn_logf)
#define FAST_LOG(x) (__builtin_amdgcn_logf(x) * 0.69314718055994530942f)   // v_log_f32
#else
#define FAST_LOG(x) __logf(x)
#endif

// One wave (32 lanes) advances 16 independent CRF chains through all T steps.
// Per step: (16x64) expA  x  (64x64) expT  via 8x v_wmma_f32_16x16x32_f16.
// WMMA layouts (CDNA5 wave32, 16-bit data):
//   A 16x32 : lane l -> row M = l&15; halfs 0..7 = K (l>>4)*8+0..7, halfs 8..15 = K 16+(l>>4)*8+0..7
//   B 32x16 : lane l -> col N = l&15; halfs 0..7 = K (l>>4)*8+0..7, halfs 8..15 = K 16+(l>>4)*8+0..7
//   C/D 16x16 f32: VGPR r, lane l -> row M = r + 8*(l>>4), col N = l&15
__global__ __launch_bounds__(32) void crf_forward(const float* __restrict__ em,
                                                  const float* __restrict__ tr,
                                                  float* __restrict__ alpha)
{
    __shared__ __align__(16) float lds_alpha[16 * Lsz];
    __shared__ __align__(16) float lds_m[16];

    const int l   = threadIdx.x;   // 0..31
    const int lo  = l & 15;
    const int hi  = l >> 4;
    const int g16 = blockIdx.x * 16;   // first batch of this wave's group

    // ---- Precompute B = exp(transition) in f16, resident in VGPRs.
    // START row (i==0) zeroed: implements scores[:,1:,START,:] = NEG (exp(NEG)=0).
    // Tr[:,START]=NEG and Tr[END,:]=NEG already give exact zeros via exp().
    v16h Bt[4][2];
#pragma unroll
    for (int jt = 0; jt < 4; ++jt) {
#pragma unroll
        for (int kc = 0; kc < 2; ++kc) {
#pragma unroll
            for (int h = 0; h < 16; ++h) {
                const int i = kc * 32 + (h < 8 ? hi * 8 + h : 16 + hi * 8 + (h - 8));
                const int j = jt * 16 + lo;
                const float w = (i == 0) ? 0.0f : FAST_EXP(tr[i * Lsz + j]);
                Bt[jt][kc][h] = (_Float16)w;
            }
        }
    }

    // ---- t = 0 : alpha0[b][j] = Tr[START][j] + em'(b,0,j)
#pragma unroll
    for (int base = 0; base < 16 * Lsz; base += 32) {
        const int q = base + l;
        const int b = q >> 6;
        const int j = q & 63;
        float e = em[(size_t)(g16 + b) * Tsz * Lsz + j];
        if (j == 0 || j == Lsz - 1) e = NEGV;            // em' mask (START/END cols)
        const float a0 = tr[j] + e;
        lds_alpha[q] = a0;
        alpha[(size_t)(g16 + b) * Tsz * Lsz + j] = a0;
    }

#pragma unroll 1
    for (int t = 1; t < Tsz; ++t) {
        // ---- issue this step's emission loads early (D-layout positions, masked).
        // Single-use data: non-temporal so it doesn't evict the hot working set.
        float emv[4][8];
#pragma unroll
        for (int jt = 0; jt < 4; ++jt) {
#pragma unroll
            for (int r = 0; r < 8; ++r) {
                const int b = r + 8 * hi;
                const int j = jt * 16 + lo;
                float e = __builtin_nontemporal_load(
                    &em[((size_t)(g16 + b) * Tsz + t) * Lsz + j]);
                if (j == 0 || j == Lsz - 1) e = NEGV;
                emv[jt][r] = e;
            }
        }
        // touch next step's rows: global_prefetch_b8
        if (t + 1 < Tsz)
            __builtin_prefetch(&em[((size_t)(g16 + lo) * Tsz + (t + 1)) * Lsz + hi * 32], 0, 3);

        // ---- per-batch row max  m[b] = max_i alpha_prev[b][i]   (lane l -> b = lo)
        float mx = -3.0e38f;
        {
            const float4* src = reinterpret_cast<const float4*>(&lds_alpha[lo * Lsz + hi * 32]);
#pragma unroll
            for (int k = 0; k < 8; ++k) {
                const float4 v = src[k];
                mx = fmaxf(mx, fmaxf(fmaxf(v.x, v.y), fmaxf(v.z, v.w)));
            }
        }
        mx = fmaxf(mx, __shfl_xor(mx, 16, 32));
        lds_m[lo] = mx;   // lanes l and l+16 write identical value

        // ---- A = exp(alpha_prev - m) in f16, A-matrix layout (lane row = lo)
        v16h A[2];
#pragma unroll
        for (int kc = 0; kc < 2; ++kc) {
            const float* row = &lds_alpha[lo * Lsz + kc * 32];
            const float4 p0 = *reinterpret_cast<const float4*>(&row[hi * 8]);
            const float4 p1 = *reinterpret_cast<const float4*>(&row[hi * 8 + 4]);
            const float4 p2 = *reinterpret_cast<const float4*>(&row[16 + hi * 8]);
            const float4 p3 = *reinterpret_cast<const float4*>(&row[16 + hi * 8 + 4]);
            const float pv[16] = { p0.x, p0.y, p0.z, p0.w,  p1.x, p1.y, p1.z, p1.w,
                                   p2.x, p2.y, p2.z, p2.w,  p3.x, p3.y, p3.z, p3.w };
#pragma unroll
            for (int h = 0; h < 16; ++h)
                A[kc][h] = (_Float16)FAST_EXP(pv[h] - mx);
        }

        // ---- broadcast m[] to D-layout rows: lane needs m[r + 8*hi], r = 0..7
        float mv[8];
        {
            const float4* mp = reinterpret_cast<const float4*>(&lds_m[hi * 8]);
            const float4 a = mp[0], b = mp[1];
            mv[0] = a.x; mv[1] = a.y; mv[2] = a.z; mv[3] = a.w;
            mv[4] = b.x; mv[5] = b.y; mv[6] = b.z; mv[7] = b.w;
        }

        // ---- 8x WMMA + log/rescale epilogue, write alpha_t
#pragma unroll
        for (int jt = 0; jt < 4; ++jt) {
            v8f acc = {};
            acc = __builtin_amdgcn_wmma_f32_16x16x32_f16(false, A[0], false, Bt[jt][0],
                                                         (short)0, acc, false, false);
            acc = __builtin_amdgcn_wmma_f32_16x16x32_f16(false, A[1], false, Bt[jt][1],
                                                         (short)0, acc, false, false);
#pragma unroll
            for (int r = 0; r < 8; ++r) {
                const float s  = fmaxf(acc[r], 1e-30f);   // all-masked columns stay ~NEG
                const float av = mv[r] + FAST_LOG(s) + emv[jt][r];
                const int b = r + 8 * hi;
                const int j = jt * 16 + lo;
                lds_alpha[b * Lsz + j] = av;
                __builtin_nontemporal_store(
                    av, &alpha[((size_t)(g16 + b) * Tsz + t) * Lsz + j]);
            }
        }
    }
}

// log_Z[b] = logsumexp_j( alpha[b, len_b-1, j] + Tr[j, END] )
__global__ __launch_bounds__(64) void crf_logz(const float* __restrict__ alpha,
                                               const float* __restrict__ tr,
                                               const int*   __restrict__ lens,
                                               float*       __restrict__ logZ)
{
    __shared__ float red[Lsz];
    const int b = blockIdx.x;
    const int j = threadIdx.x;
    const int len = lens[b];
    const float v = alpha[((size_t)b * Tsz + (len - 1)) * Lsz + j] + tr[j * Lsz + (Lsz - 1)];

    red[j] = v;
    __syncthreads();
    for (int s = 32; s > 0; s >>= 1) {
        if (j < s) red[j] = fmaxf(red[j], red[j + s]);
        __syncthreads();
    }
    const float m = red[0];
    __syncthreads();
    red[j] = FAST_EXP(v - m);
    __syncthreads();
    for (int s = 32; s > 0; s >>= 1) {
        if (j < s) red[j] += red[j + s];
        __syncthreads();
    }
    if (j == 0) logZ[b] = m + FAST_LOG(fmaxf(red[0], 1e-30f));
}

extern "C" void kernel_launch(void* const* d_in, const int* in_sizes, int n_in,
                              void* d_out, int out_size, void* d_ws, size_t ws_size,
                              hipStream_t stream)
{
    const float* em   = (const float*)d_in[0];   // (B,T,L) f32
    const float* tr   = (const float*)d_in[1];   // (L,L)   f32
    const int*   lens = (const int*)  d_in[2];   // (B,)    i32

    float* alpha = (float*)d_out;                        // (B,T,L)
    float* logZ  = alpha + (size_t)Bsz * Tsz * Lsz;      // (B,)

    crf_forward<<<dim3(4),   dim3(32), 0, stream>>>(em, tr, alpha);
    crf_logz  <<<dim3(Bsz),  dim3(Lsz), 0, stream>>>(alpha, tr, lens, logZ);
}
